// KANProjection_23321672417778
// MI455X (gfx1250) — matrix operand order repaired and checked by reading, not compiled, not run on hardware
//
#include <hip/hip_runtime.h>

// POD vector types (avoid HIP_vector_type ctors so unions stay trivial)
typedef __attribute__((ext_vector_type(16))) _Float16 v16h;
typedef __attribute__((ext_vector_type(8)))  float    v8f;
typedef __attribute__((ext_vector_type(4)))  unsigned int u32x4;
typedef __attribute__((ext_vector_type(4)))  float    f32x4;
typedef __attribute__((ext_vector_type(4)))  int      v4i;

#define IN_DIM  1024
#define OUT_DIM 2048
#define KDIM    9216          // 1024 base + 1024*8 spline
#define NROWS   8192          // 4*2048
#define NKNOT   12            // GRID + 2K + 1

// ---- CDNA5 async global->LDS path (guarded: falls back to reg staging) ----
#ifndef USE_ASYNC_LDS
#if defined(__has_builtin)
#if __has_builtin(__builtin_amdgcn_global_load_async_to_lds_b128)
#define USE_ASYNC_LDS 1
#endif
#endif
#endif
#ifndef USE_ASYNC_LDS
#define USE_ASYNC_LDS 0
#endif

union H8  { _Float16 h[8];  u32x4 u; };
union H16 { v16h v; u32x4 q[2]; };

#if USE_ASYNC_LDS
typedef __attribute__((address_space(1))) v4i* gptr_v4i;   // global v4i*
typedef __attribute__((address_space(3))) v4i* lptr_v4i;   // LDS v4i*

__device__ __forceinline__ void async_cp16(_Float16* l, const _Float16* g) {
  __builtin_amdgcn_global_load_async_to_lds_b128(
      (gptr_v4i)g, (lptr_v4i)l, 0, 0);
}
__device__ __forceinline__ void wait_async0() {
#if __has_builtin(__builtin_amdgcn_s_wait_asynccnt)
  __builtin_amdgcn_s_wait_asynccnt(0);
#else
  asm volatile("s_wait_asynccnt 0" ::: "memory");
#endif
}
#endif

// ---------------------------------------------------------------------------
// Kernel 1: per-(n,d) activations -> f16 A[N][K] row-major.
//   A[n][d]            = silu(x[n,d])                       (d < 1024)
//   A[n][1024+8d+g]    = cubic B-spline basis B[n,d,g]      (g < 8)
// ---------------------------------------------------------------------------
__global__ __launch_bounds__(256) void kan_basis(
    const float* __restrict__ x, const float* __restrict__ grid,
    _Float16* __restrict__ As)
{
  long idx = (long)blockIdx.x * 256 + threadIdx.x;   // NROWS*IN_DIM threads
  int  d   = (int)(idx & (IN_DIM - 1));
  long n   = idx >> 10;
  float xv = x[idx];

  float gk[NKNOT];
#pragma unroll
  for (int j = 0; j < NKNOT; ++j) gk[j] = grid[d * NKNOT + j];

  // Cox-de Boor, k = 3
  float B0[11];
#pragma unroll
  for (int j = 0; j < 11; ++j)
    B0[j] = (xv >= gk[j] && xv < gk[j + 1]) ? 1.0f : 0.0f;
  float B1[10];
#pragma unroll
  for (int j = 0; j < 10; ++j)
    B1[j] = (xv - gk[j]) / (gk[j + 1] - gk[j]) * B0[j]
          + (gk[j + 2] - xv) / (gk[j + 2] - gk[j + 1]) * B0[j + 1];
  float B2[9];
#pragma unroll
  for (int j = 0; j < 9; ++j)
    B2[j] = (xv - gk[j]) / (gk[j + 2] - gk[j]) * B1[j]
          + (gk[j + 3] - xv) / (gk[j + 3] - gk[j + 1]) * B1[j + 1];
  H8 r;
#pragma unroll
  for (int j = 0; j < 8; ++j) {
    float b = (xv - gk[j]) / (gk[j + 3] - gk[j]) * B2[j]
            + (gk[j + 4] - xv) / (gk[j + 4] - gk[j + 1]) * B2[j + 1];
    r.h[j] = (_Float16)b;
  }

  float s = xv / (1.0f + __expf(-xv));   // silu
  _Float16* rowp = As + n * KDIM;
  rowp[d] = (_Float16)s;
  *(u32x4*)(rowp + IN_DIM + 8 * d) = r.u;   // 16B aligned store of 8 halves
}

// ---------------------------------------------------------------------------
// Packed W layout: Wp[(k>>4)*OUT_DIM*16 + o*16 + (k&15)]  (f16)
// ---------------------------------------------------------------------------

// Kernel 2: spline rows  k = 1024 + 8d + g :  coef[d,o,g]*scale_sp[d,o]
__global__ __launch_bounds__(256) void kan_pack_spline(
    const float* __restrict__ coef, const float* __restrict__ scale_sp,
    _Float16* __restrict__ Wp)
{
  long idx = (long)blockIdx.x * 256 + threadIdx.x;   // IN_DIM*OUT_DIM threads
  int o = (int)(idx & (OUT_DIM - 1));
  int d = (int)(idx >> 11);
  float ss = scale_sp[(long)d * OUT_DIM + o];
  const f32x4* pc = (const f32x4*)(coef + ((long)d * OUT_DIM + o) * 8);
  f32x4 c0 = pc[0], c1 = pc[1];
  H8 r;
  r.h[0] = (_Float16)(c0.x * ss); r.h[1] = (_Float16)(c0.y * ss);
  r.h[2] = (_Float16)(c0.z * ss); r.h[3] = (_Float16)(c0.w * ss);
  r.h[4] = (_Float16)(c1.x * ss); r.h[5] = (_Float16)(c1.y * ss);
  r.h[6] = (_Float16)(c1.z * ss); r.h[7] = (_Float16)(c1.w * ss);
  long k = 1024 + 8L * d;                      // k&15 is 0 or 8 -> 16B aligned
  long addr = ((k >> 4) * OUT_DIM + o) * 16 + (k & 15);
  *(u32x4*)&Wp[addr] = r.u;
}

// Kernel 3: base rows k < 1024 : transpose scale_base[k][o] via LDS tile
__global__ __launch_bounds__(256) void kan_pack_base(
    const float* __restrict__ scale_base, _Float16* __restrict__ Wp)
{
  __shared__ float lds[16][257];
  int kt = blockIdx.x;            // 0..63  (16 k's each)
  int ob = blockIdx.y;            // 0..7   (256 o's each)
  int t  = threadIdx.x;
  int o  = ob * 256 + t;
#pragma unroll
  for (int j = 0; j < 16; ++j)
    lds[j][t] = scale_base[(long)(kt * 16 + j) * OUT_DIM + o];
  __syncthreads();
  union { _Float16 h[16]; u32x4 u[2]; } r;
#pragma unroll
  for (int j = 0; j < 16; ++j) r.h[j] = (_Float16)lds[j][t];
  long addr = ((long)kt * OUT_DIM + o) * 16;
  *(u32x4*)&Wp[addr]     = r.u[0];
  *(u32x4*)&Wp[addr + 8] = r.u[1];
}

// ---------------------------------------------------------------------------
// Kernel 4: C[N][O] f32 = A[N][K] f16  x  Wp (K x O) f16, f32 accumulation.
// Block = 256 threads (8 wave32), block tile 128x256, wave tile 64x64
// (4x4 WMMA fragments), double-buffered LDS, 288 k-steps.
// Staging: async global->LDS (ASYNCcnt) when available, else reg staging.
// ---------------------------------------------------------------------------
__global__ __launch_bounds__(256) void kan_gemm(
    const _Float16* __restrict__ As, const _Float16* __restrict__ Wp,
    float* __restrict__ out)
{
  constexpr int NKT = KDIM / 32;          // 288
  __shared__ _Float16 sA[2][128 * 40];    // 128 rows x 32 halves, stride 40
  __shared__ _Float16 sB[2][512 * 24];    // [kg(2)*256 + o]*24, 16 halves used

  const int t  = threadIdx.x;
  const int m0 = blockIdx.y * 128;
  const int o0 = blockIdx.x * 256;

  // ---- staging maps ----
  const int arow  = t >> 1;               // 128 rows, 2 threads/row (32B each)
  const int apart = t & 1;
  const _Float16* gA = As + (long)(m0 + arow) * KDIM + apart * 16;
  const int laoff = arow * 40 + apart * 16;

  const _Float16* gBu[4];                 // 4 x 16B units per thread
  int lboff[4];
#pragma unroll
  for (int j = 0; j < 4; ++j) {
    int u = t + 256 * j;                  // 1024 x 16B units = 16KB B tile
    int c = u >> 9, rem = u & 511, o = rem >> 1, h = rem & 1;
    gBu[j]   = Wp + ((long)c * OUT_DIM + (o0 + o)) * 16 + h * 8;
    lboff[j] = (c * 256 + o) * 24 + h * 8;
  }
  const long bstep = (long)2 * OUT_DIM * 16;   // halves per k-step

  // ---- wave/lane map ----
  const int lane = t & 31;
  const int w    = t >> 5;
  const int l16  = lane & 15;
  const int lhi  = lane >> 4;
  const int mlb  = (w & 1) * 64;
  const int olb  = (w >> 1) * 64;

  v8f acc[4][4] = {};

#if USE_ASYNC_LDS
  auto issue = [&](int kt, int buf) {
    const _Float16* pA = gA + (long)kt * 32;
    async_cp16(&sA[buf][laoff],     pA);
    async_cp16(&sA[buf][laoff + 8], pA + 8);
#pragma unroll
    for (int j = 0; j < 4; ++j)
      async_cp16(&sB[buf][lboff[j]], gBu[j] + (long)kt * bstep);
  };
  issue(0, 0);
  wait_async0();
  __syncthreads();
#else
  u32x4 ra0, ra1, rb[4];
  auto gload = [&](int kt) {
    const u32x4* pA = (const u32x4*)(gA + (long)kt * 32);
    ra0 = pA[0]; ra1 = pA[1];
#pragma unroll
    for (int j = 0; j < 4; ++j)
      rb[j] = *(const u32x4*)(gBu[j] + (long)kt * bstep);
  };
  auto stlds = [&](int buf) {
    *(u32x4*)&sA[buf][laoff]     = ra0;
    *(u32x4*)&sA[buf][laoff + 8] = ra1;
#pragma unroll
    for (int j = 0; j < 4; ++j)
      *(u32x4*)&sB[buf][lboff[j]] = rb[j];
  };
  gload(0); stlds(0);
  gload(1);
  __syncthreads();
#endif

  for (int kt = 0; kt < NKT; ++kt) {
    const int buf = kt & 1;

#if USE_ASYNC_LDS
    if (kt + 1 < NKT) issue(kt + 1, buf ^ 1);   // prev barrier makes this safe
#endif
    if (kt + 4 < NKT) {                         // L2 look-ahead
      __builtin_prefetch(gA + (long)(kt + 4) * 32, 0, 1);
      __builtin_prefetch(gBu[0] + (long)(kt + 4) * bstep, 0, 1);
    }

    // B fragments: lane = column, 16 contiguous K halves starting at lhi*16
    H16 bf[4];
#pragma unroll
    for (int nt = 0; nt < 4; ++nt) {
      const _Float16* pb = &sB[buf][(lhi * 256 + olb + nt * 16 + l16) * 24];
      bf[nt].q[0] = *(const u32x4*)pb;
      bf[nt].q[1] = *(const u32x4*)(pb + 8);
    }
#pragma unroll
    for (int mt = 0; mt < 4; ++mt) {
      // A fragment: 16 contiguous K halves starting at lhi*16 (matches B)
      const _Float16* pa = &sA[buf][(mlb + mt * 16 + l16) * 40 + lhi * 16];
      H16 a;
      a.q[0] = *(const u32x4*)pa;
      a.q[1] = *(const u32x4*)(pa + 8);
#pragma unroll
      for (int nt = 0; nt < 4; ++nt)
        acc[mt][nt] = __builtin_amdgcn_wmma_f32_16x16x32_f16(
            false, a.v, false, bf[nt].v, (short)0, acc[mt][nt], false, false);
    }

    if (kt + 1 < NKT) {
#if USE_ASYNC_LDS
      wait_async0();        // own async stores to LDS landed
      __syncthreads();      // publish tile to all waves
#else
      __syncthreads();
      stlds(buf ^ 1);
      if (kt + 2 < NKT) gload(kt + 2);
      __syncthreads();
#endif
    }
  }

  // epilogue: C layout -> lane col = l16, row = lhi*8 + vgpr
#pragma unroll
  for (int mt = 0; mt < 4; ++mt)
#pragma unroll
    for (int nt = 0; nt < 4; ++nt) {
      const int row = m0 + mlb + mt * 16 + lhi * 8;
      const int col = o0 + olb + nt * 16 + l16;
      float* po = out + (long)row * OUT_DIM + col;
#pragma unroll
      for (int v = 0; v < 8; ++v)
        po[(long)v * OUT_DIM] = acc[mt][nt][v];
    }
}

// ---------------------------------------------------------------------------
extern "C" void kernel_launch(void* const* d_in, const int* in_sizes, int n_in,
                              void* d_out, int out_size, void* d_ws, size_t ws_size,
                              hipStream_t stream) {
  const float* x          = (const float*)d_in[0];
  const float* grid       = (const float*)d_in[1];
  const float* coef       = (const float*)d_in[2];
  const float* scale_base = (const float*)d_in[3];
  const float* scale_sp   = (const float*)d_in[4];
  float* out = (float*)d_out;

  // workspace: As (f16, N x K) then Wp (f16, K/16 x O x 16). ~189 MB total.
  _Float16* As = (_Float16*)d_ws;
  _Float16* Wp = (_Float16*)((char*)d_ws + (size_t)NROWS * KDIM * 2);

  kan_basis<<<(NROWS * IN_DIM) / 256, 256, 0, stream>>>(x, grid, As);
  kan_pack_spline<<<(IN_DIM * OUT_DIM) / 256, 256, 0, stream>>>(coef, scale_sp, Wp);
  {
    dim3 g(IN_DIM / 16, OUT_DIM / 256);
    kan_pack_base<<<g, 256, 0, stream>>>(scale_base, Wp);
  }
  {
    dim3 g(OUT_DIM / 256, NROWS / 128);
    kan_gemm<<<g, 256, 0, stream>>>(As, Wp, out);
  }
}
